// JetMoeParallelExperts_25546465477250
// MI455X (gfx1250) — matrix operand order repaired and checked by reading, not compiled, not run on hardware
//
#include <hip/hip_runtime.h>
#include <stdint.h>

typedef __attribute__((ext_vector_type(16))) __bf16 v16bf;
typedef __attribute__((ext_vector_type(8)))  float  v8f;

#define NE   8
#define MPE  32768
#define NN   512
#define KK   512
#define BM   128
#define BN   64
#define BK   32
#define LDA  40            // padded bf16 row stride: 80 bytes = 5 x 16B (aligned, conflict-free)
#define KT   (KK / BK)     // 16 K-steps

struct Q2 { uint4 a, b; };   // 32 bytes == v16bf

static __device__ __forceinline__ uint32_t fbits(float f) {
  union { float f; uint32_t u; } v; v.f = f; return v.u;
}
static __device__ __forceinline__ float ffrom(uint32_t u) {
  union { uint32_t u; float f; } v; v.u = u; return v.f;
}

// Truncation-based fp32 -> (bf16 hi, bf16 lo) split. x - hi is exact in f32,
// so |x - hi - lo| ~ 2^-15 * |x|. Cheap: and/sub/shift-or per element.
static __device__ __forceinline__ void cvt4(float4 x, uint2* hi, uint2* lo) {
  uint32_t u0 = fbits(x.x), u1 = fbits(x.y), u2 = fbits(x.z), u3 = fbits(x.w);
  uint32_t h0 = u0 & 0xffff0000u, h1 = u1 & 0xffff0000u;
  uint32_t h2 = u2 & 0xffff0000u, h3 = u3 & 0xffff0000u;
  hi->x = (u0 >> 16) | h1;
  hi->y = (u2 >> 16) | h3;
  uint32_t r0 = fbits(x.x - ffrom(h0)), r1 = fbits(x.y - ffrom(h1));
  uint32_t r2 = fbits(x.z - ffrom(h2)), r3 = fbits(x.w - ffrom(h3));
  lo->x = (r0 >> 16) | (r1 & 0xffff0000u);
  lo->y = (r2 >> 16) | (r3 & 0xffff0000u);
}

// 16 contiguous bf16 (32B), 16B-aligned -> v16bf fragment (two ds_load_b128)
static __device__ __forceinline__ v16bf ldfrag16(const uint16_t* p) {
  Q2 q;
  q.a = *(const uint4*)(p + 0);
  q.b = *(const uint4*)(p + 8);
  return __builtin_bit_cast(v16bf, q);
}

__global__ __launch_bounds__(256)
void moe_grouped_gemm_bf16x3(const float* __restrict__ X,
                             const float* __restrict__ W,
                             float* __restrict__ O) {
  // A rows stored in K order (no swizzle), so each half-wave's
  // 16-element WMMA A-fragment is contiguous & 16B aligned. B rows stored in K order
  // (half-wave fragments are naturally contiguous at byte offsets 0 / 32).
  __shared__ alignas(16) uint16_t Ah[2][BM][LDA];
  __shared__ alignas(16) uint16_t Al[2][BM][LDA];
  __shared__ alignas(16) uint16_t Bh[2][BN][LDA];
  __shared__ alignas(16) uint16_t Bl[2][BN][LDA];

  const int tid  = threadIdx.x;
  const int lane = tid & 31;
  const int wid  = tid >> 5;
  const int wm   = wid & 3;   // 4 wave slots along M (32 rows each)
  const int wn   = wid >> 2;  // 2 wave slots along N (32 cols each)

  const int  gn    = blockIdx.x;            // 0..7
  const int  gm    = blockIdx.y;            // 0..2047
  const long mbase = (long)gm * BM;
  const int  e     = gm >> 8;               // 32768/128 = 256 M-blocks per expert
  const int  nbase = gn * BN;

  const float* Ab = X + mbase * KK;
  const float* Wb = W + ((long)e * NN + nbase) * (long)KK;

  v8f acc[2][2];
  #pragma unroll
  for (int i = 0; i < 2; ++i)
    #pragma unroll
    for (int j = 0; j < 2; ++j) acc[i][j] = 0.0f;

  auto load_stage = [&](int kt, int buf) {
    const int kof = kt * BK;
    #pragma unroll
    for (int p = 0; p < 4; ++p) {            // A: 128 rows x 32 K (fp32)
      int idx = tid + p * 256;
      int r = idx >> 3, c4 = idx & 7;
      // K order (no swizzle), matching B's contiguous fragments
      int c4s = c4;
      float4 v = *(const float4*)(Ab + (long)r * KK + kof + c4 * 4);
      uint2 hi, lo; cvt4(v, &hi, &lo);
      *(uint2*)&Ah[buf][r][c4s * 4] = hi;
      *(uint2*)&Al[buf][r][c4s * 4] = lo;
    }
    #pragma unroll
    for (int p = 0; p < 2; ++p) {            // B: 64 rows(N) x 32 K (fp32)
      int idx = tid + p * 256;
      int r = idx >> 3, c4 = idx & 7;
      float4 v = *(const float4*)(Wb + (long)r * KK + kof + c4 * 4);
      uint2 hi, lo; cvt4(v, &hi, &lo);
      *(uint2*)&Bh[buf][r][c4 * 4] = hi;
      *(uint2*)&Bl[buf][r][c4 * 4] = lo;
    }
  };

  load_stage(0, 0);
  __syncthreads();

  const int rA = lane & 15;
  const int kh = (lane >> 4) * 16;   // half-wave fragment offset (elements)

  for (int kt = 0; kt < KT; ++kt) {
    const int buf = kt & 1;
    if (kt + 1 < KT) load_stage(kt + 1, buf ^ 1);

    v16bf ah[2], al[2], bh[2], bl[2];
    #pragma unroll
    for (int mt = 0; mt < 2; ++mt) {
      const int row = wm * 32 + mt * 16 + rA;
      ah[mt] = ldfrag16(&Ah[buf][row][kh]);
      al[mt] = ldfrag16(&Al[buf][row][kh]);
    }
    #pragma unroll
    for (int nt = 0; nt < 2; ++nt) {
      const int col = wn * 32 + nt * 16 + rA;
      bh[nt] = ldfrag16(&Bh[buf][col][kh]);
      bl[nt] = ldfrag16(&Bl[buf][col][kh]);
    }

    #pragma unroll
    for (int mt = 0; mt < 2; ++mt)
      #pragma unroll
      for (int nt = 0; nt < 2; ++nt) {
        acc[mt][nt] = __builtin_amdgcn_wmma_f32_16x16x32_bf16(
            false, ah[mt], false, bh[nt], (short)0, acc[mt][nt], false, false);
        acc[mt][nt] = __builtin_amdgcn_wmma_f32_16x16x32_bf16(
            false, ah[mt], false, bl[nt], (short)0, acc[mt][nt], false, false);
        acc[mt][nt] = __builtin_amdgcn_wmma_f32_16x16x32_bf16(
            false, al[mt], false, bh[nt], (short)0, acc[mt][nt], false, false);
      }

    __syncthreads();
  }

  // Epilogue: 16x16 f32 C layout — VGPR r: lanes 0-15 -> M=r, lanes 16-31 -> M=r+8.
  // Non-temporal stores: output is streamed once, keep L2 free for X tile re-reads.
  const int cn = lane & 15;
  const int mh = (lane >> 4) * 8;
  #pragma unroll
  for (int mt = 0; mt < 2; ++mt)
    #pragma unroll
    for (int nt = 0; nt < 2; ++nt) {
      const long row0 = mbase + wm * 32 + mt * 16 + mh;
      const int  col  = nbase + wn * 32 + nt * 16 + cn;
      float* o = O + row0 * NN + col;
      #pragma unroll
      for (int r = 0; r < 8; ++r)
        __builtin_nontemporal_store(acc[mt][nt][r], o + (long)r * NN);
    }
}

extern "C" void kernel_launch(void* const* d_in, const int* in_sizes, int n_in,
                              void* d_out, int out_size, void* d_ws, size_t ws_size,
                              hipStream_t stream) {
  const float* X = (const float*)d_in[0];   // [262144, 512] fp32, sorted by expert
  const float* W = (const float*)d_in[1];   // [8, 512, 512] fp32
  float* O = (float*)d_out;                 // [262144, 512] fp32
  dim3 grid(NN / BN, (NE * MPE) / BM);      // (8, 2048)
  moe_grouped_gemm_bf16x3<<<grid, 256, 0, stream>>>(X, W, O);
}
